// ImprovedSimpleEdgeGNN_81793357185761
// MI455X (gfx1250) — matrix operand order repaired and checked by reading, compiled, not run on hardware
//
#include <hip/hip_runtime.h>
#include <stdint.h>
#include <stddef.h>

// ---------------- problem constants ----------------
#define NN   65536      // nodes
#define EE   131072     // edges
#define GG   2048       // graphs
#define DIN  768
#define DH   128
#define HC   512        // HEADS*DH
#define DPOS 63
#define KIN  832        // DIN + DPOS padded to mult of 32 (831 -> 832)
#define NCLS 5

typedef __attribute__((ext_vector_type(16))) __bf16 v16bf;
typedef __attribute__((ext_vector_type(8)))  float  v8f;
typedef __attribute__((ext_vector_type(4)))  uint32_t v4u;
typedef __attribute__((ext_vector_type(8)))  int      v8i;
typedef __attribute__((ext_vector_type(4)))  int      v4i;

#if defined(__has_builtin)
#if __has_builtin(__builtin_amdgcn_tensor_load_to_lds) && __has_builtin(__builtin_amdgcn_s_wait_tensorcnt)
#define HAVE_TDM 1
#endif
#endif

__device__ __forceinline__ uint16_t f2bf_bits(float f) {
  uint32_t u = __float_as_uint(f);
  uint32_t r = u + 0x7fffu + ((u >> 16) & 1u);   // round-to-nearest-even
  return (uint16_t)(r >> 16);
}
__device__ __forceinline__ int ord_key(float f) {
  int s = __float_as_int(f);
  return (s >= 0) ? s : (s ^ 0x7fffffff);
}
__device__ __forceinline__ float ord_dec(int s) {
  return __int_as_float((s >= 0) ? s : (s ^ 0x7fffffff));
}

// ---------------- WMMA GEMM, TDM-staged B, 1x4 N-register-blocking ----------------
// C[M,N] = A[M,K]bf16 * Bt[N,K]bf16 (+bias)(+C if beta)(relu)
// Block (8 waves) owns a 128-row M panel x 64-col N strip. The 64xK B strip is
// DMA'd once into LDS by the Tensor Data Mover; each wave then runs a 16x64
// output strip: 1 global A fragment + 4 LDS B fragments feed 4 WMMAs per K-step.
// Requires M % 128 == 0, N % 64 == 0, K % 32 == 0 (true for all call sites).
__global__ void gemm_bf16_wmma_kernel(const uint16_t* __restrict__ A, int lda,
                                      const uint16_t* __restrict__ Bt, int ldb,
                                      const float* __restrict__ bias,
                                      float* __restrict__ C, int ldc,
                                      int M, int Ncols, int K, int beta, int relu)
{
  extern __shared__ uint16_t ldsB[];           // [64][K] bf16 strip of Bt
  int w    = threadIdx.x >> 5;
  int lane = threadIdx.x & 31;
  int groupsN = Ncols >> 6;
  int blockM  = blockIdx.x / groupsN;
  int tg      = blockIdx.x - blockM * groupsN;
  int tm      = blockM * 8 + w;

  // ---- stage B strip: Bt[tg*64 .. tg*64+63][0..K) -> LDS ----
#ifdef HAVE_TDM
  if (w == 0) {
    uint64_t gaddr = (uint64_t)(uintptr_t)(Bt + (size_t)tg * 64 * ldb);
    uint32_t laddr = (uint32_t)(uintptr_t)ldsB;
    v4u g0;
    g0[0] = 1u;                                            // count=1 (valid user D#)
    g0[1] = laddr;                                         // lds_addr [63:32]
    g0[2] = (uint32_t)gaddr;                               // global_addr[31:0]
    g0[3] = (uint32_t)((gaddr >> 32) & 0x01FFFFFFu) | (2u << 30); // addr[56:32] | type=2
    v8i g1;
    g1[0] = (int)(1u << 16);                               // data_size=1 -> 2 bytes/elem
    g1[1] = (int)(((uint32_t)K & 0xffffu) << 16);          // tensor_dim0[15:0] @ [63:48]
    g1[2] = (int)(((uint32_t)K >> 16) |
                  (((uint32_t)Ncols & 0xffffu) << 16));    // dim0[31:16] | dim1[15:0]
    g1[3] = (int)(((uint32_t)Ncols >> 16) |
                  ((uint32_t)K << 16));                    // dim1[31:16] | tile_dim0=K
    g1[4] = 64;                                            // tile_dim1=64, tile_dim2=0
    g1[5] = ldb;                                           // tensor_dim0_stride[31:0]
    g1[6] = 0;                                             // stride hi | dim1_stride lo
    g1[7] = 0;
    v4i gz4 = (v4i){0, 0, 0, 0};                           // 2D tensor: groups 2/3 unused
    v8i gz8 = (v8i){0, 0, 0, 0, 0, 0, 0, 0};
    __builtin_amdgcn_tensor_load_to_lds(g0, g1, gz4, gz4, gz8, 0);
    __builtin_amdgcn_s_wait_tensorcnt(0);
  }
  __syncthreads();
#else
  for (int i = threadIdx.x; i < 64 * (K >> 3); i += 256) {
    int r = i / (K >> 3), cc = i - r * (K >> 3);
    *(uint4*)(&ldsB[(size_t)r * K + cc * 8]) =
        *(const uint4*)(Bt + (size_t)(tg * 64 + r) * ldb + cc * 8);
  }
  __syncthreads();
#endif

  int half = lane >> 4;
  int l15  = lane & 15;
  const uint16_t* Aptr = A + (size_t)(tm * 16 + l15) * lda;
  const uint16_t* Lp0 = ldsB + (size_t)( 0 + l15) * K;
  const uint16_t* Lp1 = ldsB + (size_t)(16 + l15) * K;
  const uint16_t* Lp2 = ldsB + (size_t)(32 + l15) * K;
  const uint16_t* Lp3 = ldsB + (size_t)(48 + l15) * K;

  v8f acc0, acc1, acc2, acc3;
  if (beta) {
#pragma unroll
    for (int vv = 0; vv < 8; ++vv) {
      size_t rowoff = (size_t)(tm * 16 + vv + 8 * half) * ldc + tg * 64 + l15;
      acc0[vv] = C[rowoff +  0];
      acc1[vv] = C[rowoff + 16];
      acc2[vv] = C[rowoff + 32];
      acc3[vv] = C[rowoff + 48];
    }
  } else {
    acc0 = (v8f){0.f,0.f,0.f,0.f,0.f,0.f,0.f,0.f};
    acc1 = acc0; acc2 = acc0; acc3 = acc0;
  }

  for (int k0 = 0; k0 < K; k0 += 32) {
    union { v16bf v; uint32_t u[8]; } af, b0, b1, b2, b3;
    __builtin_prefetch(Aptr + k0 + 128, 0, 0);   // stream A ahead into GL2
#pragma unroll
    for (int vv = 0; vv < 8; ++vv) {
      // A 16x32 bf16, contiguous k order used here: lanes0-15 K=0..15 (2/VGPR), lanes16-31 K=16..31
      int ka = k0 + half * 16 + vv * 2;
      af.u[vv] = *(const uint32_t*)(Aptr + ka);
      // B 32x16 bf16, contiguous k order used here: lanes0-15 K=0..15 (2/VGPR), lanes16-31 K=16..31
      int kb = k0 + half * 16 + vv * 2;
      b0.u[vv] = *(const uint32_t*)(Lp0 + kb);
      b1.u[vv] = *(const uint32_t*)(Lp1 + kb);
      b2.u[vv] = *(const uint32_t*)(Lp2 + kb);
      b3.u[vv] = *(const uint32_t*)(Lp3 + kb);
    }
    acc0 = __builtin_amdgcn_wmma_f32_16x16x32_bf16(false, af.v, false, b0.v, (short)0, acc0, false, false);
    acc1 = __builtin_amdgcn_wmma_f32_16x16x32_bf16(false, af.v, false, b1.v, (short)0, acc1, false, false);
    acc2 = __builtin_amdgcn_wmma_f32_16x16x32_bf16(false, af.v, false, b2.v, (short)0, acc2, false, false);
    acc3 = __builtin_amdgcn_wmma_f32_16x16x32_bf16(false, af.v, false, b3.v, (short)0, acc3, false, false);
  }

#pragma unroll
  for (int vv = 0; vv < 8; ++vv) {
    int m = tm * 16 + vv + 8 * half;
    int nbase = tg * 64 + l15;
    float v0 = acc0[vv], v1 = acc1[vv], v2 = acc2[vv], v3 = acc3[vv];
    if (bias) {
      v0 += bias[nbase + 0]; v1 += bias[nbase + 16];
      v2 += bias[nbase + 32]; v3 += bias[nbase + 48];
    }
    if (relu) {
      v0 = fmaxf(v0, 0.f); v1 = fmaxf(v1, 0.f);
      v2 = fmaxf(v2, 0.f); v3 = fmaxf(v3, 0.f);
    }
    size_t rowoff = (size_t)m * ldc + nbase;
    C[rowoff +  0] = v0;
    C[rowoff + 16] = v1;
    C[rowoff + 32] = v2;
    C[rowoff + 48] = v3;
  }
}

// ---------------- small helper kernels ----------------
__global__ void wtrans_kernel(const float* __restrict__ src, uint16_t* __restrict__ dst,
                              int K, int Nc, int Kpad) {
  int i = blockIdx.x * blockDim.x + threadIdx.x;
  if (i >= Nc * Kpad) return;
  int n = i / Kpad, k = i - n * Kpad;
  float v = (k < K) ? src[(size_t)k * Nc + n] : 0.f;
  dst[i] = f2bf_bits(v);
}

__global__ void cvt_bf16_kernel(const float* __restrict__ src, uint16_t* __restrict__ dst, int n) {
  int i = blockIdx.x * blockDim.x + threadIdx.x;
  if (i < n) dst[i] = f2bf_bits(src[i]);
}

__global__ void fill_i32_kernel(int* p, int val, int n) {
  int i = blockIdx.x * blockDim.x + threadIdx.x;
  if (i < n) p[i] = val;
}

__global__ void pos_concat_kernel(const float* __restrict__ x, const int* __restrict__ pos_ids,
                                  const float* __restrict__ pos_emb, uint16_t* __restrict__ Xc) {
  int n = blockIdx.x;
  __shared__ int ids[8];
  if (threadIdx.x < 8) ids[threadIdx.x] = pos_ids[n * 8 + threadIdx.x];
  __syncthreads();
  int cntI = 0;
#pragma unroll
  for (int j = 0; j < 8; ++j) cntI += (ids[j] != 0);
  float cnt = fmaxf((float)cntI, 1.f);
  for (int c = threadIdx.x; c < KIN; c += blockDim.x) {
    float val;
    if (c < DIN) val = x[(size_t)n * DIN + c];
    else if (c < DIN + DPOS) {
      int d = c - DIN; float s = 0.f;
#pragma unroll
      for (int j = 0; j < 8; ++j) { int id = ids[j]; if (id) s += pos_emb[id * DPOS + d]; }
      val = s / cnt;
    } else val = 0.f;
    Xc[(size_t)n * KIN + c] = f2bf_bits(val);
  }
}

__global__ void copy_cvt_kernel(const float* __restrict__ h, float* __restrict__ hres,
                                uint16_t* __restrict__ hbf, int total) {
  int i = blockIdx.x * blockDim.x + threadIdx.x;
  if (i >= total) return;
  float v = h[i];
  if (hres) hres[i] = v;
  hbf[i] = f2bf_bits(v);
}

__global__ void add_res_cvt_kernel(float* __restrict__ h, const float* __restrict__ hres,
                                   uint16_t* __restrict__ hbf, int total) {
  int i = blockIdx.x * blockDim.x + threadIdx.x;
  if (i >= total) return;
  float v = h[i] + hres[i];
  h[i] = v;
  hbf[i] = f2bf_bits(v);
}

__global__ void edge_emb_gather_kernel(const int* __restrict__ eattr, const float* __restrict__ emb,
                                       uint16_t* __restrict__ out, int total) {
  int i = blockIdx.x * blockDim.x + threadIdx.x;
  if (i >= total) return;
  int e = i >> 7, c = i & 127;
  out[i] = f2bf_bits(emb[eattr[e] * DH + c]);
}

__global__ void edge_sum_kernel(const int* __restrict__ src, const int* __restrict__ dst,
                                const float* __restrict__ ef, float* __restrict__ ssum,
                                float* __restrict__ dsum, int total) {
  int i = blockIdx.x * blockDim.x + threadIdx.x;
  if (i >= total) return;
  int e = i >> 7, c = i & 127;
  float v = ef[i];
  atomicAdd(&ssum[(size_t)src[e] * DH + c], v);
  atomicAdd(&dsum[(size_t)dst[e] * DH + c], v);
}

__global__ void edge_cnt_kernel(const int* __restrict__ src, const int* __restrict__ dst,
                                float* __restrict__ scnt, float* __restrict__ dcnt, int E) {
  int e = blockIdx.x * blockDim.x + threadIdx.x;
  if (e >= E) return;
  atomicAdd(&scnt[src[e]], 1.f);
  atomicAdd(&dcnt[dst[e]], 1.f);
}

__global__ void agg_concat_kernel(const float* __restrict__ ssum, const float* __restrict__ scnt,
                                  const float* __restrict__ dsum, const float* __restrict__ dcnt,
                                  uint16_t* __restrict__ out, int total) {
  int i = blockIdx.x * blockDim.x + threadIdx.x;
  if (i >= total) return;
  int n = i >> 8, c = i & 255;
  float v = (c < DH) ? ssum[(size_t)n * DH + c] / fmaxf(scnt[n], 1.f)
                     : dsum[(size_t)n * DH + (c - DH)] / fmaxf(dcnt[n], 1.f);
  out[i] = f2bf_bits(v);
}

// alpha[e,h] = dot(q[dst], k[src]+e) / sqrt(128); one wave / edge, 8 lanes / head
__global__ void alpha_kernel(const int* __restrict__ src, const int* __restrict__ dst,
                             const float* __restrict__ q, const float* __restrict__ k,
                             const float* __restrict__ e, float* __restrict__ alpha, int E) {
  int wave = (int)((blockIdx.x * blockDim.x + threadIdx.x) >> 5);
  int lane = threadIdx.x & 31;
  if (wave >= E) return;
  int s = src[wave], d = dst[wave];
  int head = lane >> 3, sub = lane & 7;
  const float* qp = q + (size_t)d * HC + head * DH + sub * 16;
  const float* kp = k + (size_t)s * HC + head * DH + sub * 16;
  const float* ep = e + (size_t)wave * HC + head * DH + sub * 16;
  float acc = 0.f;
#pragma unroll
  for (int i = 0; i < 16; ++i) acc += qp[i] * (kp[i] + ep[i]);
  for (int off = 4; off; off >>= 1) acc += __shfl_xor(acc, off, 32);
  if (sub == 0) alpha[(size_t)wave * 4 + head] = acc * 0.08838834764831845f;
}

__global__ void seg_max_kernel(const float* __restrict__ alpha, const int* __restrict__ dst,
                               int* __restrict__ mx, int total) {
  int i = blockIdx.x * blockDim.x + threadIdx.x;
  if (i >= total) return;
  int e = i >> 2;
  atomicMax(&mx[dst[e] * 4 + (i & 3)], ord_key(alpha[i]));
}

__global__ void seg_exp_kernel(float* __restrict__ alpha, const int* __restrict__ dst,
                               const int* __restrict__ mx, float* __restrict__ ssum, int total) {
  int i = blockIdx.x * blockDim.x + threadIdx.x;
  if (i >= total) return;
  int e = i >> 2, h = i & 3;
  int enc = mx[dst[e] * 4 + h];
  float m = (enc == (int)0x80000000) ? 0.f : ord_dec(enc);
  float ev = __expf(alpha[i] - m);
  alpha[i] = ev;
  atomicAdd(&ssum[dst[e] * 4 + h], ev);
}

__global__ void msg_kernel(const int* __restrict__ src, const int* __restrict__ dst,
                           const float* __restrict__ v, const float* __restrict__ e,
                           const float* __restrict__ alpha, const float* __restrict__ ssum,
                           float* __restrict__ out, int total) {
  int i = blockIdx.x * blockDim.x + threadIdx.x;
  if (i >= total) return;
  int edge = i >> 9, c = i & 511;
  int h = c >> 7;
  int s = src[edge], d = dst[edge];
  float w = alpha[(size_t)edge * 4 + h] / (ssum[d * 4 + h] + 1e-16f);
  atomicAdd(&out[(size_t)d * HC + c], (v[(size_t)s * HC + c] + e[i]) * w);
}

__global__ void headmean_kernel(const float* __restrict__ in512, float* __restrict__ out128, int total) {
  int i = blockIdx.x * blockDim.x + threadIdx.x;
  if (i >= total) return;
  int n = i >> 7, c = i & 127;
  const float* p = in512 + (size_t)n * HC + c;
  out128[i] = 0.25f * (p[0] + p[128] + p[256] + p[384]);
}

// h = h + relu(LN(outbuf; g,b)) + agg ; also refresh bf16 copy. 128 threads / node.
__global__ void ln_update_kernel(float* __restrict__ h, uint16_t* __restrict__ hbf,
                                 const float* __restrict__ outbuf,
                                 const float* __restrict__ g, const float* __restrict__ b,
                                 const float* __restrict__ agg) {
  int n = blockIdx.x, c = threadIdx.x;
  size_t idx = (size_t)n * DH + c;
  float v = outbuf[idx];
  __shared__ float s1[DH], s2[DH];
  s1[c] = v; s2[c] = v * v;
  __syncthreads();
  for (int off = 64; off; off >>= 1) {
    if (c < off) { s1[c] += s1[c + off]; s2[c] += s2[c + off]; }
    __syncthreads();
  }
  float mean = s1[0] * (1.f / DH);
  float var  = s2[0] * (1.f / DH) - mean * mean;
  float ln = (v - mean) * rsqrtf(var + 1e-5f) * g[c] + b[c];
  float hn = h[idx] + fmaxf(ln, 0.f) + (agg ? agg[idx] : 0.f);
  h[idx] = hn;
  hbf[idx] = f2bf_bits(hn);
}

__global__ void ln_graph_kernel(const float* __restrict__ in, float* __restrict__ out) {
  int n = blockIdx.x, c = threadIdx.x;
  size_t idx = (size_t)n * DH + c;
  float v = in[idx];
  __shared__ float s1[DH], s2[DH];
  s1[c] = v; s2[c] = v * v;
  __syncthreads();
  for (int off = 64; off; off >>= 1) {
    if (c < off) { s1[c] += s1[c + off]; s2[c] += s2[c + off]; }
    __syncthreads();
  }
  float mean = s1[0] * (1.f / DH);
  float var  = s2[0] * (1.f / DH) - mean * mean;
  out[idx] = (v - mean) * rsqrtf(var + 1e-5f);
}

__global__ void gate2_kernel(const float* __restrict__ g1, const float* __restrict__ w,
                             const float* __restrict__ b2, float* __restrict__ gatev, int N) {
  int wave = (int)((blockIdx.x * blockDim.x + threadIdx.x) >> 5);
  int lane = threadIdx.x & 31;
  if (wave >= N) return;
  float acc = 0.f;
  for (int i = lane; i < DH; i += 32) acc += g1[(size_t)wave * DH + i] * w[i];
  for (int off = 16; off; off >>= 1) acc += __shfl_xor(acc, off, 32);
  if (lane == 0) gatev[wave] = acc + b2[0];
}

__global__ void pool_max_kernel(const float* __restrict__ gatev, const int* __restrict__ batch,
                                int* __restrict__ pmax, int N) {
  int n = blockIdx.x * blockDim.x + threadIdx.x;
  if (n < N) atomicMax(&pmax[batch[n]], ord_key(gatev[n]));
}
__global__ void pool_exp_kernel(const float* __restrict__ gatev, const int* __restrict__ batch,
                                const int* __restrict__ pmax, float* __restrict__ expg,
                                float* __restrict__ psum, int N) {
  int n = blockIdx.x * blockDim.x + threadIdx.x;
  if (n >= N) return;
  int enc = pmax[batch[n]];
  float m = (enc == (int)0x80000000) ? 0.f : ord_dec(enc);
  float ev = __expf(gatev[n] - m);
  expg[n] = ev;
  atomicAdd(&psum[batch[n]], ev);
}
__global__ void pool_accum_kernel(const float* __restrict__ expg, const float* __restrict__ psum,
                                  const int* __restrict__ batch, const float* __restrict__ h,
                                  float* __restrict__ pooled, int total) {
  int i = blockIdx.x * blockDim.x + threadIdx.x;
  if (i >= total) return;
  int n = i >> 7, c = i & 127;
  int g = batch[n];
  float w = expg[n] / (psum[g] + 1e-16f);
  atomicAdd(&pooled[(size_t)g * DH + c], w * h[i]);
}

__global__ void lin_small_kernel(const float* __restrict__ X, const float* __restrict__ W,
                                 const float* __restrict__ bias, float* __restrict__ out,
                                 int G, int Out) {
  int i = blockIdx.x * blockDim.x + threadIdx.x;
  if (i >= G * Out) return;
  int g = i / Out, j = i - g * Out;
  float acc = bias[j];
  for (int k = 0; k < DH; ++k) acc += X[(size_t)g * DH + k] * W[k * Out + j];
  out[i] = acc;
}

__global__ void coral_kernel(const float* __restrict__ logits, const int* __restrict__ y,
                             float* __restrict__ acc, int G) {
  int g = blockIdx.x * blockDim.x + threadIdx.x;
  if (g >= G) return;
  float s = 0.f; int yy = y[g];
#pragma unroll
  for (int j = 0; j < NCLS - 1; ++j) {
    float x = logits[g * (NCLS - 1) + j];
    float ls = (x >= 0.f) ? -log1pf(__expf(-x)) : (x - log1pf(__expf(x)));
    float lvl = (yy > j) ? 1.f : 0.f;
    s += ls * lvl + (ls - x) * (1.f - lvl);
  }
  atomicAdd(acc, -s);
}

__global__ void kappa_stats_kernel(const float* __restrict__ lw, const int* __restrict__ y,
                                   float* __restrict__ O, float* __restrict__ psum,
                                   float* __restrict__ ycnt, int G) {
  int g = blockIdx.x * blockDim.x + threadIdx.x;
  if (g >= G) return;
  float v[NCLS]; float m = -1e30f;
#pragma unroll
  for (int c = 0; c < NCLS; ++c) { v[c] = lw[g * NCLS + c]; m = fmaxf(m, v[c]); }
  float s = 0.f;
#pragma unroll
  for (int c = 0; c < NCLS; ++c) { v[c] = __expf(v[c] - m); s += v[c]; }
  float inv = 1.f / s;
  float s2 = 0.f;
#pragma unroll
  for (int c = 0; c < NCLS; ++c) { v[c] *= inv; s2 += v[c]; }
  float inv2 = 1.f / (s2 + 1e-10f);
  int yy = y[g];
#pragma unroll
  for (int c = 0; c < NCLS; ++c) {
    float p = v[c] * inv2;
    atomicAdd(&O[c * NCLS + yy], p);
    atomicAdd(&psum[c], p);
  }
  atomicAdd(&ycnt[yy], 1.f);
}

__global__ void final_loss_kernel(const float* __restrict__ O, const float* __restrict__ psum,
                                  const float* __restrict__ ycnt, const float* __restrict__ coral_acc,
                                  float* __restrict__ out0, int G) {
  float num = 0.f, den = 0.f;
  for (int i = 0; i < NCLS; ++i)
    for (int j = 0; j < NCLS; ++j) {
      float w = (float)((i - j) * (i - j)) / 16.f;   // max weight = (4-0)^2
      num += w * O[i * NCLS + j];
      den += w * psum[i] * ycnt[j] / (float)G;
    }
  float kappa = 1.f - num / (den + 1e-10f);
  float wkl = logf(1.f - kappa + 1e-10f);
  out0[0] = 0.5f * (coral_acc[0] / (float)G) + 0.5f * wkl;
}

// ---------------- host orchestration ----------------
static inline int cdiv(long a, long b) { return (int)((a + b - 1) / b); }

static void launch_gemm(const uint16_t* A, int lda, const uint16_t* Bt, int ldb,
                        const float* bias, float* C, int ldc, int M, int N, int K,
                        int beta, int relu, hipStream_t s) {
  int blocks = (M / 128) * (N / 64);          // 128-row panel x 64-col strip per block
  size_t shbytes = (size_t)64 * K * 2;        // LDS B strip
  gemm_bf16_wmma_kernel<<<blocks, 256, shbytes, s>>>(A, lda, Bt, ldb, bias, C, ldc,
                                                     M, N, K, beta, relu);
}

extern "C" void kernel_launch(void* const* d_in, const int* in_sizes, int n_in,
                              void* d_out, int out_size, void* d_ws, size_t ws_size,
                              hipStream_t stream) {
  (void)in_sizes; (void)n_in; (void)out_size; (void)ws_size;

  const float* x        = (const float*)d_in[0];
  const int*   esrc     = (const int*)d_in[1];
  const int*   edst     = esrc + EE;
  const int*   eattr    = (const int*)d_in[2];
  const int*   pos_ids  = (const int*)d_in[3];
  const int*   batch    = (const int*)d_in[4];
  const int*   y        = (const int*)d_in[5];
  const float* pos_emb  = (const float*)d_in[6];
  const float* edge_emb = (const float*)d_in[7];
  const float* ep_w1 = (const float*)d_in[8],  * ep_b1 = (const float*)d_in[9];
  const float* ep_w2 = (const float*)d_in[10], * ep_b2 = (const float*)d_in[11];
  const float* np_w1 = (const float*)d_in[12], * np_b1 = (const float*)d_in[13];
  const float* np_w2 = (const float*)d_in[14], * np_b2 = (const float*)d_in[15];
  const float* fuse_w = (const float*)d_in[16], * fuse_b = (const float*)d_in[17];
  const float* gate_w1 = (const float*)d_in[51], * gate_b1 = (const float*)d_in[52];
  const float* gate_w2 = (const float*)d_in[53], * gate_b2 = (const float*)d_in[54];
  const float* fc_w  = (const float*)d_in[55], * fc_b  = (const float*)d_in[56];
  const float* fcw_w = (const float*)d_in[57], * fcw_b = (const float*)d_in[58];

  float* out = (float*)d_out;           // [0]=loss, [1..8192]=logits

  // workspace carve-out (deterministic)
  char* wp = (char*)d_ws;
  auto alloc = [&](size_t bytes) -> void* {
    void* r = (void*)wp; wp += (bytes + 255) & ~(size_t)255; return r;
  };
  uint16_t* Xc     = (uint16_t*)alloc((size_t)NN * KIN * 2);
  float*    h      = (float*)alloc((size_t)NN * DH * 4);
  float*    h_res  = (float*)alloc((size_t)NN * DH * 4);
  float*    htmp   = (float*)alloc((size_t)NN * DH * 4);
  uint16_t* h_bf   = (uint16_t*)alloc((size_t)NN * DH * 2);
  uint16_t* tmp_bf = (uint16_t*)alloc((size_t)EE * DH * 2);
  float*    ef     = (float*)alloc((size_t)EE * DH * 4);
  uint16_t* ef_bf  = (uint16_t*)alloc((size_t)EE * DH * 2);
  float*    q      = (float*)alloc((size_t)NN * HC * 4);
  float*    k      = (float*)alloc((size_t)NN * HC * 4);
  float*    v      = (float*)alloc((size_t)NN * HC * 4);
  float*    e512   = (float*)alloc((size_t)EE * HC * 4);
  float*    msgout = (float*)alloc((size_t)NN * HC * 4);
  float*    out128 = (float*)alloc((size_t)NN * DH * 4);
  float*    alphaB = (float*)alloc((size_t)EE * 4 * 4);
  int*      mx     = (int*)alloc((size_t)NN * 4 * 4);
  float*    ssumA  = (float*)alloc((size_t)NN * 4 * 4);
  float*    s_sum  = (float*)alloc((size_t)NN * DH * 4);
  float*    d_sum  = (float*)alloc((size_t)NN * DH * 4);
  float*    s_cnt  = (float*)alloc((size_t)NN * 4);
  float*    d_cnt  = (float*)alloc((size_t)NN * 4);
  uint16_t* Aagg   = (uint16_t*)alloc((size_t)NN * 256 * 2);
  float*    agg    = (float*)alloc((size_t)NN * DH * 4);
  float*    gatev  = (float*)alloc((size_t)NN * 4);
  float*    expg   = (float*)alloc((size_t)NN * 4);
  int*      pmax   = (int*)alloc((size_t)GG * 4);
  float*    psump  = (float*)alloc((size_t)GG * 4);
  float*    pooled = (float*)alloc((size_t)GG * DH * 4);
  float*    pooledL= (float*)alloc((size_t)GG * DH * 4);
  float*    wklbuf = (float*)alloc((size_t)GG * NCLS * 4);
  float*    Omat   = (float*)alloc(25 * 4);
  float*    psum5  = (float*)alloc(5 * 4);
  float*    ycnt   = (float*)alloc(5 * 4);
  float*    coralA = (float*)alloc(4);
  uint16_t* np_w1t = (uint16_t*)alloc((size_t)DH * KIN * 2);
  uint16_t* np_w2t = (uint16_t*)alloc((size_t)DH * DH * 2);
  uint16_t* ep_w1t = (uint16_t*)alloc((size_t)DH * DH * 2);
  uint16_t* ep_w2t = (uint16_t*)alloc((size_t)DH * DH * 2);
  uint16_t* fuse_wt= (uint16_t*)alloc((size_t)DH * 256 * 2);
  uint16_t* gw1t   = (uint16_t*)alloc((size_t)DH * DH * 2);
  uint16_t* wqt[3], * wkt[3], * wvt[3], * wet[3], * wst[3];
  for (int l = 0; l < 3; ++l) {
    wqt[l] = (uint16_t*)alloc((size_t)HC * DH * 2);
    wkt[l] = (uint16_t*)alloc((size_t)HC * DH * 2);
    wvt[l] = (uint16_t*)alloc((size_t)HC * DH * 2);
    wet[l] = (uint16_t*)alloc((size_t)HC * DH * 2);
    wst[l] = (uint16_t*)alloc((size_t)DH * DH * 2);
  }

  // ---- weight transposition / bf16 conversion ----
  wtrans_kernel<<<cdiv((long)DH * KIN, 256), 256, 0, stream>>>(np_w1, np_w1t, DIN + DPOS, DH, KIN);
  wtrans_kernel<<<cdiv((long)DH * DH, 256), 256, 0, stream>>>(np_w2, np_w2t, DH, DH, DH);
  wtrans_kernel<<<cdiv((long)DH * DH, 256), 256, 0, stream>>>(ep_w1, ep_w1t, DH, DH, DH);
  wtrans_kernel<<<cdiv((long)DH * DH, 256), 256, 0, stream>>>(ep_w2, ep_w2t, DH, DH, DH);
  wtrans_kernel<<<cdiv((long)DH * 256, 256), 256, 0, stream>>>(fuse_w, fuse_wt, 256, DH, 256);
  wtrans_kernel<<<cdiv((long)DH * DH, 256), 256, 0, stream>>>(gate_w1, gw1t, DH, DH, DH);
  for (int l = 0; l < 3; ++l) {
    int cb = 18 + l * 11;
    wtrans_kernel<<<cdiv((long)HC * DH, 256), 256, 0, stream>>>((const float*)d_in[cb + 0], wqt[l], DH, HC, DH);
    wtrans_kernel<<<cdiv((long)HC * DH, 256), 256, 0, stream>>>((const float*)d_in[cb + 2], wkt[l], DH, HC, DH);
    wtrans_kernel<<<cdiv((long)HC * DH, 256), 256, 0, stream>>>((const float*)d_in[cb + 4], wvt[l], DH, HC, DH);
    wtrans_kernel<<<cdiv((long)HC * DH, 256), 256, 0, stream>>>((const float*)d_in[cb + 6], wet[l], DH, HC, DH);
    wtrans_kernel<<<cdiv((long)DH * DH, 256), 256, 0, stream>>>((const float*)d_in[cb + 7], wst[l], DH, DH, DH);
  }

  // ---- node features: pos embedding + concat (bf16) -> node MLP (WMMA) ----
  pos_concat_kernel<<<NN, 256, 0, stream>>>(x, pos_ids, pos_emb, Xc);
  launch_gemm(Xc, KIN, np_w1t, KIN, np_b1, htmp, DH, NN, DH, KIN, 0, 1, stream);
  cvt_bf16_kernel<<<cdiv((long)NN * DH, 256), 256, 0, stream>>>(htmp, tmp_bf, NN * DH);
  launch_gemm(tmp_bf, DH, np_w2t, DH, np_b2, h, DH, NN, DH, DH, 0, 1, stream);
  copy_cvt_kernel<<<cdiv((long)NN * DH, 256), 256, 0, stream>>>(h, h_res, h_bf, NN * DH);

  // ---- edge features: embed -> 2-layer MLP (WMMA) ----
  edge_emb_gather_kernel<<<cdiv((long)EE * DH, 256), 256, 0, stream>>>(eattr, edge_emb, tmp_bf, EE * DH);
  launch_gemm(tmp_bf, DH, ep_w1t, DH, ep_b1, ef, DH, EE, DH, DH, 0, 1, stream);
  cvt_bf16_kernel<<<cdiv((long)EE * DH, 256), 256, 0, stream>>>(ef, tmp_bf, EE * DH);
  launch_gemm(tmp_bf, DH, ep_w2t, DH, ep_b2, ef, DH, EE, DH, DH, 0, 0, stream);
  cvt_bf16_kernel<<<cdiv((long)EE * DH, 256), 256, 0, stream>>>(ef, ef_bf, EE * DH);

  // ---- loop-invariant edge aggregation (same every layer) ----
  (void)hipMemsetAsync(s_sum, 0, (size_t)NN * DH * 4, stream);
  (void)hipMemsetAsync(d_sum, 0, (size_t)NN * DH * 4, stream);
  (void)hipMemsetAsync(s_cnt, 0, (size_t)NN * 4, stream);
  (void)hipMemsetAsync(d_cnt, 0, (size_t)NN * 4, stream);
  edge_sum_kernel<<<cdiv((long)EE * DH, 256), 256, 0, stream>>>(esrc, edst, ef, s_sum, d_sum, EE * DH);
  edge_cnt_kernel<<<cdiv(EE, 256), 256, 0, stream>>>(esrc, edst, s_cnt, d_cnt, EE);
  agg_concat_kernel<<<cdiv((long)NN * 256, 256), 256, 0, stream>>>(s_sum, s_cnt, d_sum, d_cnt, Aagg, NN * 256);
  launch_gemm(Aagg, 256, fuse_wt, 256, fuse_b, agg, DH, NN, DH, 256, 0, 0, stream);

  // ---- 3 transformer-conv layers ----
  for (int l = 0; l < 3; ++l) {
    int cb = 18 + l * 11;
    const float* bq = (const float*)d_in[cb + 1];
    const float* bk = (const float*)d_in[cb + 3];
    const float* bv = (const float*)d_in[cb + 5];
    const float* bs = (const float*)d_in[cb + 8];
    const float* lng = (const float*)d_in[cb + 9];
    const float* lnb = (const float*)d_in[cb + 10];

    launch_gemm(h_bf, DH, wqt[l], DH, bq, q, HC, NN, HC, DH, 0, 0, stream);
    launch_gemm(h_bf, DH, wkt[l], DH, bk, k, HC, NN, HC, DH, 0, 0, stream);
    launch_gemm(h_bf, DH, wvt[l], DH, bv, v, HC, NN, HC, DH, 0, 0, stream);
    launch_gemm(ef_bf, DH, wet[l], DH, nullptr, e512, HC, EE, HC, DH, 0, 0, stream);

    alpha_kernel<<<cdiv((long)EE * 32, 256), 256, 0, stream>>>(esrc, edst, q, k, e512, alphaB, EE);
    fill_i32_kernel<<<cdiv((long)NN * 4, 256), 256, 0, stream>>>(mx, (int)0x80000000, NN * 4);
    seg_max_kernel<<<cdiv((long)EE * 4, 256), 256, 0, stream>>>(alphaB, edst, mx, EE * 4);
    (void)hipMemsetAsync(ssumA, 0, (size_t)NN * 4 * 4, stream);
    seg_exp_kernel<<<cdiv((long)EE * 4, 256), 256, 0, stream>>>(alphaB, edst, mx, ssumA, EE * 4);
    (void)hipMemsetAsync(msgout, 0, (size_t)NN * HC * 4, stream);
    msg_kernel<<<cdiv((long)EE * HC, 256), 256, 0, stream>>>(esrc, edst, v, e512, alphaB, ssumA, msgout, EE * HC);
    headmean_kernel<<<cdiv((long)NN * DH, 256), 256, 0, stream>>>(msgout, out128, NN * DH);
    // skip connection: out128 += h @ ws + bs (beta=1)
    launch_gemm(h_bf, DH, wst[l], DH, bs, out128, DH, NN, DH, DH, 1, 0, stream);
    // h = h + relu(LN(out128)) + agg ; refresh bf16 copy
    ln_update_kernel<<<NN, DH, 0, stream>>>(h, h_bf, out128, lng, lnb, agg);
  }

  // ---- global residual + attentional pooling ----
  add_res_cvt_kernel<<<cdiv((long)NN * DH, 256), 256, 0, stream>>>(h, h_res, h_bf, NN * DH);
  launch_gemm(h_bf, DH, gw1t, DH, gate_b1, htmp, DH, NN, DH, DH, 0, 1, stream);
  gate2_kernel<<<cdiv((long)NN * 32, 256), 256, 0, stream>>>(htmp, gate_w2, gate_b2, gatev, NN);
  fill_i32_kernel<<<cdiv(GG, 256), 256, 0, stream>>>(pmax, (int)0x80000000, GG);
  pool_max_kernel<<<cdiv(NN, 256), 256, 0, stream>>>(gatev, batch, pmax, NN);
  (void)hipMemsetAsync(psump, 0, (size_t)GG * 4, stream);
  pool_exp_kernel<<<cdiv(NN, 256), 256, 0, stream>>>(gatev, batch, pmax, expg, psump, NN);
  (void)hipMemsetAsync(pooled, 0, (size_t)GG * DH * 4, stream);
  pool_accum_kernel<<<cdiv((long)NN * DH, 256), 256, 0, stream>>>(expg, psump, batch, h, pooled, NN * DH);
  ln_graph_kernel<<<GG, DH, 0, stream>>>(pooled, pooledL);

  // ---- heads + losses ----
  lin_small_kernel<<<cdiv((long)GG * (NCLS - 1), 256), 256, 0, stream>>>(pooledL, fc_w, fc_b, out + 1, GG, NCLS - 1);
  lin_small_kernel<<<cdiv((long)GG * NCLS, 256), 256, 0, stream>>>(pooledL, fcw_w, fcw_b, wklbuf, GG, NCLS);
  (void)hipMemsetAsync(coralA, 0, 4, stream);
  (void)hipMemsetAsync(Omat, 0, 25 * 4, stream);
  (void)hipMemsetAsync(psum5, 0, 5 * 4, stream);
  (void)hipMemsetAsync(ycnt, 0, 5 * 4, stream);
  coral_kernel<<<cdiv(GG, 256), 256, 0, stream>>>(out + 1, y, coralA, GG);
  kappa_stats_kernel<<<cdiv(GG, 256), 256, 0, stream>>>(wklbuf, y, Omat, psum5, ycnt, GG);
  final_loss_kernel<<<1, 1, 0, stream>>>(Omat, psum5, ycnt, coralA, out, GG);
}
